// MetaR_14053132992849
// MI455X (gfx1250) — matrix operand order repaired and checked by reading, not compiled, not run on hardware
//
#include <hip/hip_runtime.h>
#include <cstdint>
#include <cstddef>

// ---------------------------------------------------------------------------
// CDNA5 (gfx1250) MetaR pipeline: bf16 WMMA GEMMs + gather/LSTM/meta kernels.
// ---------------------------------------------------------------------------

typedef __bf16 bf16_t;
typedef bf16_t v16bf __attribute__((ext_vector_type(16)));
typedef float  v8f   __attribute__((ext_vector_type(8)));

#define DEV static __device__ __forceinline__

DEV unsigned short f2bf(float f) {
  unsigned u = __float_as_uint(f);
  u += 0x7fffu + ((u >> 16) & 1u);
  return (unsigned short)(u >> 16);
}
DEV float bf2f(unsigned short h) { return __uint_as_float(((unsigned)h) << 16); }
DEV float sigf(float x) { return 1.f / (1.f + __expf(-x)); }

// A fragment (16x32 bf16, row-major source, ld multiple of 16, 16B-aligned):
// lane half h = lane>>4, r = lane&15 : elems 0..7 -> K = k0 + 8h + i,
//                                      elems 8..15 -> K = k0 + 16 + 8h + (i-8)
DEV v16bf load_a_frag(const unsigned short* A, int lda, int mbase, int k0) {
  const int lane = threadIdx.x & 31;
  const int r = lane & 15, kb = (lane >> 4) * 8;
  const uint4* q = reinterpret_cast<const uint4*>(A + (size_t)(mbase + r) * lda + k0 + kb);
  union { uint4 u[2]; v16bf v; } f;
  f.u[0] = q[0];   // K = k0+kb .. +7
  f.u[1] = q[2];   // K = k0+kb+16 .. +23
  return f.v;
}

// B fragment (32x16 bf16) from BT row-major (N x K) = weight (out,in) layout:
// lane half h : elems 0..7 -> K = k0 + 8h + i, elems 8..15 -> K = k0 + 16 + 8h + (i-8), N = lane&15
DEV v16bf load_b_frag(const unsigned short* BT, int ldb, int nbase, int k0) {
  const int lane = threadIdx.x & 31;
  const int n = nbase + (lane & 15), kb = (lane >> 4) * 8;
  const uint4* q = reinterpret_cast<const uint4*>(BT + (size_t)n * ldb + k0 + kb);
  union { uint4 u[2]; v16bf v; } f;
  f.u[0] = q[0];
  f.u[1] = q[2];
  return f.v;
}

DEV v8f wmma_bf16(v16bf a, v16bf b, v8f c) {
  return __builtin_amdgcn_wmma_f32_16x16x32_bf16(false, a, false, b, (short)0, c,
                                                 false, false);
}

// ---------------------------------------------------------------------------
// Utility kernels
// ---------------------------------------------------------------------------
__global__ void zero_kernel(uint4* p, size_t n) {
  size_t i = (size_t)blockIdx.x * blockDim.x + threadIdx.x;
  size_t stride = (size_t)gridDim.x * blockDim.x;
  uint4 z = make_uint4(0, 0, 0, 0);
  for (; i < n; i += stride) p[i] = z;
}

__global__ void conv_bf16(const float* __restrict__ src, unsigned short* __restrict__ dst,
                          int rows, int cols, int dstld) {
  int i = blockIdx.x * blockDim.x + threadIdx.x;
  if (i >= rows * cols) return;
  int r = i / cols, c = i % cols;
  dst[(size_t)r * dstld + c] = f2bf(src[i]);
}

// ---------------------------------------------------------------------------
// Generic bf16 WMMA GEMM: C[M x Npad] = A[M x K]@BT^T (+ Cadd) (+ bias0/bias1)
//  - A: bf16 row-major (lda mult of 16), M multiple of 64 (grid.x = M/64)
//  - BT: bf16, (Npad rows x K), i.e. weight (out,in) layout
//  - K multiple of 32 (padded with zeros), Npad multiple of 16
//  - each wave computes a 64(M) x 16(N) strip: 4 accumulators, B-frag reuse x4
//    -> 1.25 fragment loads per WMMA and 4 independent XDL dependency chains
//  - 8 waves/block, grid.y = ceil(Npad/128)
// ---------------------------------------------------------------------------
__global__ __launch_bounds__(256) void gemm_bf16_wmma(
    const unsigned short* __restrict__ A, int lda,
    const unsigned short* __restrict__ BT, int ldb,
    float* __restrict__ C, int ldc,
    const float* __restrict__ Cadd, int ldca,
    const float* __restrict__ bias0, const float* __restrict__ bias1,
    int Npad, int Nvalid, int K) {
  const int lane = threadIdx.x & 31;
  const int wv = threadIdx.x >> 5;
  const int nt = blockIdx.y * 8 + wv;
  if ((nt + 1) * 16 > Npad) return;  // wave-uniform: EXEC stays all-ones for WMMA
  const int mbase = blockIdx.x * 64;
  const int nbase = nt * 16;
  v8f acc[4];
#pragma unroll
  for (int mb = 0; mb < 4; ++mb) acc[mb] = (v8f){0.f, 0.f, 0.f, 0.f, 0.f, 0.f, 0.f, 0.f};
  for (int k0 = 0; k0 < K; k0 += 32) {
    v16bf b = load_b_frag(BT, ldb, nbase, k0);
#pragma unroll
    for (int mb = 0; mb < 4; ++mb) {
      v16bf a = load_a_frag(A, lda, mbase + mb * 16, k0);
      acc[mb] = wmma_bf16(a, b, acc[mb]);
    }
  }
  const int half = lane >> 4, l15 = lane & 15;
  const int n = nbase + l15;
  float badd = 0.f;
  if (n < Nvalid) {
    if (bias0) badd += bias0[n];
    if (bias1) badd += bias1[n];
  }
#pragma unroll
  for (int mb = 0; mb < 4; ++mb) {
#pragma unroll
    for (int j = 0; j < 8; ++j) {
      int m = mbase + mb * 16 + j + 8 * half;
      float v = acc[mb][j] + badd;
      if (Cadd) v += Cadd[(size_t)m * ldca + n];
      C[(size_t)m * ldc + n] = v;
    }
  }
}

// ---------------------------------------------------------------------------
// Neighbor encoder: one block per node (2560 nodes), 8 waves.
// GEMM out[200x100] = cat[200x200] @ gcn_w^T via WMMA (cat staged in LDS bf16,
// K chunked 128+96), then leaky_relu -> attention softmax -> gate -> x0 bf16.
// ---------------------------------------------------------------------------
__global__ __launch_bounds__(256) void encoder_kernel(
    const int* __restrict__ left_conn, const int* __restrict__ right_conn,
    const float* __restrict__ emb,
    const unsigned short* __restrict__ gcnw,  // 112 x 224 bf16 (zero padded)
    const float* __restrict__ gcn_wb, const float* __restrict__ gcn_b,
    const float* __restrict__ attn_w,
    const float* __restrict__ gate_w, const float* __restrict__ gate_wb,
    const float* __restrict__ gate_b,
    unsigned short* __restrict__ x0)          // 1280 x 224 bf16
{
  __shared__ unsigned short catA[208 * 128];  // reused as out (208x112 bf16)
  __shared__ float sh_log[208];
  __shared__ float sh_oa[112];
  __shared__ float red[256];
  __shared__ float sh_scal[1];

  const int node = blockIdx.x;
  const int t = node / 512;           // FEW index
  const int side = (node / 256) & 1;  // 0=left, 1=right
  const int bb = node & 255;
  const int* conn = (side ? right_conn : left_conn) + (size_t)(t * 256 + bb) * 200 * 3;

  const int tid = threadIdx.x;
  const int lane = tid & 31;
  const int wv = tid >> 5;
  const int half = lane >> 4;
  const int l15 = lane & 15;

  v8f acc[12];
#pragma unroll
  for (int q = 0; q < 12; ++q) acc[q] = (v8f){0.f, 0.f, 0.f, 0.f, 0.f, 0.f, 0.f, 0.f};

  for (int chunk = 0; chunk < 2; ++chunk) {
    // stage gathered cat (rel_e | ent_e) as bf16, rows/cols zero-padded
    for (int idx = tid; idx < 208 * 128; idx += 256) {
      int m = idx >> 7, c = idx & 127;
      int kg = chunk * 128 + c;
      unsigned short v = 0;
      if (m < 200 && kg < 200) {
        int slot = (kg < 100) ? 1 : 2;
        int sym = conn[m * 3 + slot];
        int dd = (kg < 100) ? kg : (kg - 100);
        v = f2bf(emb[(size_t)sym * 100 + dd]);
      }
      catA[idx] = v;
    }
    __syncthreads();
    const int nks = (chunk == 0) ? 4 : 3;  // K: 128 then 96 (of 224 padded)
#pragma unroll
    for (int q = 0; q < 12; ++q) {
      int tile = wv + q * 8;   // wave-uniform guard
      if (tile < 91) {
        int mt = tile / 7, nt = tile % 7;
        for (int ks = 0; ks < nks; ++ks) {
          v16bf a = load_a_frag(catA, 128, mt * 16, ks * 32);
          v16bf b = load_b_frag(gcnw, 224, nt * 16, chunk * 128 + ks * 32);
          acc[q] = wmma_bf16(a, b, acc[q]);
        }
      }
    }
    __syncthreads();
  }

  // out = leaky_relu(acc + bias) -> store bf16 into LDS (stride 112)
  unsigned short* outL = catA;
#pragma unroll
  for (int q = 0; q < 12; ++q) {
    int tile = wv + q * 8;
    if (tile < 91) {
      int mt = tile / 7, nt = tile % 7;
      int n = nt * 16 + l15;
      float bias = (n < 100) ? (gcn_wb[n] + gcn_b[n]) : 0.f;
#pragma unroll
      for (int j = 0; j < 8; ++j) {
        int m = mt * 16 + j + 8 * half;
        float v = acc[q][j] + bias;
        v = (v > 0.f) ? v : 0.01f * v;
        outL[m * 112 + n] = f2bf(v);
      }
    }
  }
  __syncthreads();

  // attention logits a[m] = out[m,:] . attn_w
  for (int m = tid; m < 200; m += 256) {
    float a = 0.f;
    for (int n = 0; n < 100; ++n) a += bf2f(outL[m * 112 + n]) * attn_w[n];
    sh_log[m] = a;
  }
  __syncthreads();
  // softmax over m
  float v = (tid < 200) ? sh_log[tid] : -1e30f;
  red[tid] = v; __syncthreads();
  for (int s = 128; s > 0; s >>= 1) { if (tid < s) red[tid] = fmaxf(red[tid], red[tid + s]); __syncthreads(); }
  float mx = red[0]; __syncthreads();
  float e = (tid < 200) ? __expf(sh_log[tid] - mx) : 0.f;
  red[tid] = e; __syncthreads();
  for (int s = 128; s > 0; s >>= 1) { if (tid < s) red[tid] += red[tid + s]; __syncthreads(); }
  float denom = red[0]; __syncthreads();
  if (tid < 200) sh_log[tid] = e / denom;
  __syncthreads();
  // out_attn[n] = sum_m out[m,n] * p[m]
  if (tid < 100) {
    float s = 0.f;
    for (int m = 0; m < 200; ++m) s += bf2f(outL[m * 112 + tid]) * sh_log[m];
    sh_oa[tid] = s;
  }
  __syncthreads();
  // gate
  float gv = (tid < 100) ? sh_oa[tid] * gate_w[tid] : 0.f;
  red[tid] = gv; __syncthreads();
  for (int s = 128; s > 0; s >>= 1) { if (tid < s) red[tid] += red[tid + s]; __syncthreads(); }
  if (tid == 0) sh_scal[0] = sigf(red[0] + gate_wb[0] + gate_b[0]);
  __syncthreads();
  float g = sh_scal[0];
  if (tid < 100) {
    int self_sym = conn[0];  // conn[m=0, slot 0]
    float se = emb[(size_t)self_sym * 100 + tid];
    float r = sh_oa[tid] * g + se * (1.f - g);
    x0[(size_t)(t * 256 + bb) * 224 + side * 100 + tid] = f2bf(r);
  }
}

// ---------------------------------------------------------------------------
// LSTM cell elementwise update (gates already in G = x@Wih^T + h@Whh^T + b)
// ---------------------------------------------------------------------------
__global__ __launch_bounds__(256) void lstm_cell(
    const float* __restrict__ G, float* __restrict__ c,
    unsigned short* __restrict__ hbf,
    unsigned short* __restrict__ ybf, float* __restrict__ yf,
    float* __restrict__ finals, int dirOff, int tpos) {
  int idx = blockIdx.x * 256 + threadIdx.x;
  if (idx >= 256 * 450) return;
  int b = idx / 450, hh = idx % 450;
  const float* g = G + (size_t)b * 1808;
  float iv = sigf(g[hh]);
  float fv = sigf(g[450 + hh]);
  float gg = tanhf(g[900 + hh]);
  float ov = sigf(g[1350 + hh]);
  float cv = fv * c[b * 480 + hh] + iv * gg;
  float hv = ov * tanhf(cv);
  c[b * 480 + hh] = cv;
  hbf[b * 480 + hh] = f2bf(hv);
  if (ybf) ybf[(size_t)(tpos * 256 + b) * 928 + dirOff + hh] = f2bf(hv);
  if (yf)  yf[(size_t)(tpos * 256 + b) * 900 + dirOff + hh] = hv;
  if (finals) finals[b * 450 + hh] = hv;
}

// ---------------------------------------------------------------------------
// Temporal attention + context (hidden = finals flat-reshaped, per reference)
// ---------------------------------------------------------------------------
__global__ __launch_bounds__(256) void attn_kernel(
    const float* __restrict__ y1,       // (t,b,900)
    const float* __restrict__ finals,   // flat (4*256*450) == (b,900,2) reshape
    unsigned short* __restrict__ ctx)   // (256,1824) bf16
{
  __shared__ float red[256];
  __shared__ float sdots[10];
  __shared__ float aw[10];
  int b = blockIdx.x, tid = threadIdx.x;
  float dots[10];
#pragma unroll
  for (int p = 0; p < 10; ++p) dots[p] = 0.f;
  for (int h = tid; h < 900; h += 256) {
    float h0 = finals[(size_t)b * 1800 + h * 2 + 0];
    float h1 = finals[(size_t)b * 1800 + h * 2 + 1];
#pragma unroll
    for (int t = 0; t < 5; ++t) {
      float o = y1[(size_t)(t * 256 + b) * 900 + h];
      dots[t * 2 + 0] += o * h0;
      dots[t * 2 + 1] += o * h1;
    }
  }
  for (int p = 0; p < 10; ++p) {
    red[tid] = dots[p]; __syncthreads();
    for (int s = 128; s > 0; s >>= 1) { if (tid < s) red[tid] += red[tid + s]; __syncthreads(); }
    if (tid == 0) sdots[p] = red[0];
    __syncthreads();
  }
  if (tid == 0) {
    for (int l = 0; l < 2; ++l) {
      float mx = -1e30f;
      for (int t = 0; t < 5; ++t) mx = fmaxf(mx, sdots[t * 2 + l]);
      float s = 0.f, ex[5];
      for (int t = 0; t < 5; ++t) { ex[t] = __expf(sdots[t * 2 + l] - mx); s += ex[t]; }
      for (int t = 0; t < 5; ++t) aw[t * 2 + l] = ex[t] / s;
    }
  }
  __syncthreads();
  for (int h = tid; h < 900; h += 256) {
    float c0 = 0.f, c1 = 0.f;
#pragma unroll
    for (int t = 0; t < 5; ++t) {
      float o = y1[(size_t)(t * 256 + b) * 900 + h];
      c0 += o * aw[t * 2 + 0];
      c1 += o * aw[t * 2 + 1];
    }
    ctx[(size_t)b * 1824 + h * 2 + 0] = f2bf(c0);
    ctx[(size_t)b * 1824 + h * 2 + 1] = f2bf(c1);
  }
}

// ---------------------------------------------------------------------------
// Meta-gradient + query scoring. One block per batch element.
// grad_rel[b] = 1/(B*FEW) * sum_j 1[1 - p_j + n_j > 0] (v_p/||v_p|| - v_n/||v_n||)
// ---------------------------------------------------------------------------
__global__ __launch_bounds__(256) void meta_kernel(
    const float* __restrict__ support, const float* __restrict__ supneg,
    const float* __restrict__ query, const float* __restrict__ negative,
    const float* __restrict__ normv, const float* __restrict__ relbuf,  // ld 112
    float* __restrict__ out) {
  __shared__ float sh_grad[100];
  __shared__ float sh_rq[100], sh_nq[100];
  int b = blockIdx.x, tid = threadIdx.x;
  int lane = tid & 31, wv = tid >> 5;
  if (tid < 100) sh_grad[tid] = 0.f;
  __syncthreads();
  if (wv < 5) {
    int j = wv;
    float e1[4], e2[4], f1[4], f2[4], nv[4], rl[4];
#pragma unroll
    for (int s = 0; s < 4; ++s) {
      int d = lane + 32 * s;
      bool ok = d < 100;
      e1[s] = ok ? support[((size_t)(b * 5 + j) * 2 + 0) * 100 + d] : 0.f;
      e2[s] = ok ? support[((size_t)(b * 5 + j) * 2 + 1) * 100 + d] : 0.f;
      f1[s] = ok ? supneg[((size_t)(b * 5 + j) * 2 + 0) * 100 + d] : 0.f;
      f2[s] = ok ? supneg[((size_t)(b * 5 + j) * 2 + 1) * 100 + d] : 0.f;
      nv[s] = ok ? normv[(size_t)b * 100 + d] : 0.f;
      rl[s] = ok ? relbuf[(size_t)b * 112 + d] : 0.f;
    }
    float d1 = 0, d2 = 0, d3 = 0, d4 = 0;
#pragma unroll
    for (int s = 0; s < 4; ++s) { d1 += e1[s] * nv[s]; d2 += e2[s] * nv[s]; d3 += f1[s] * nv[s]; d4 += f2[s] * nv[s]; }
#pragma unroll
    for (int m = 16; m > 0; m >>= 1) {
      d1 += __shfl_xor(d1, m, 32); d2 += __shfl_xor(d2, m, 32);
      d3 += __shfl_xor(d3, m, 32); d4 += __shfl_xor(d4, m, 32);
    }
    float vp[4], vn[4], sp = 0.f, sn = 0.f;
#pragma unroll
    for (int s = 0; s < 4; ++s) {
      vp[s] = (e1[s] - d1 * nv[s]) + rl[s] - (e2[s] - d2 * nv[s]);
      vn[s] = (f1[s] - d3 * nv[s]) + rl[s] - (f2[s] - d4 * nv[s]);
      sp += vp[s] * vp[s]; sn += vn[s] * vn[s];
    }
#pragma unroll
    for (int m = 16; m > 0; m >>= 1) { sp += __shfl_xor(sp, m, 32); sn += __shfl_xor(sn, m, 32); }
    float normp = sqrtf(sp), normn = sqrtf(sn);
    if ((1.0f + normp - normn) > 0.f) {  // 1 - p + n, p=-normp, n=-normn
      float ip = (normp > 0.f) ? 1.f / normp : 0.f;
      float in_ = (normn > 0.f) ? 1.f / normn : 0.f;
#pragma unroll
      for (int s = 0; s < 4; ++s) {
        int d = lane + 32 * s;
        if (d < 100) atomicAdd(&sh_grad[d], (vp[s] * ip - vn[s] * in_) * (1.f / 1280.f));
      }
    }
  }
  __syncthreads();
  if (tid < 100) {
    float g = sh_grad[tid];
    sh_rq[tid] = relbuf[(size_t)b * 112 + tid] - 5.f * g;   // BETA = 5
    sh_nq[tid] = normv[(size_t)b * 100 + tid] - 5.f * g;
  }
  __syncthreads();
  for (int i = wv; i < 522; i += 8) {   // NQ + NN = 10 + 512, wave-uniform loop
    const float* bp = (i < 10) ? (query + (size_t)(b * 10 + i) * 2 * 100)
                               : (negative + (size_t)(b * 512 + (i - 10)) * 2 * 100);
    float e1[4], e2[4], nv[4], rq[4];
#pragma unroll
    for (int s = 0; s < 4; ++s) {
      int d = lane + 32 * s;
      bool ok = d < 100;
      e1[s] = ok ? bp[d] : 0.f;
      e2[s] = ok ? bp[100 + d] : 0.f;
      nv[s] = ok ? sh_nq[d] : 0.f;
      rq[s] = ok ? sh_rq[d] : 0.f;
    }
    float d1 = 0.f, d2 = 0.f;
#pragma unroll
    for (int s = 0; s < 4; ++s) { d1 += e1[s] * nv[s]; d2 += e2[s] * nv[s]; }
#pragma unroll
    for (int m = 16; m > 0; m >>= 1) { d1 += __shfl_xor(d1, m, 32); d2 += __shfl_xor(d2, m, 32); }
    float ss = 0.f;
#pragma unroll
    for (int s = 0; s < 4; ++s) {
      float v = (e1[s] - d1 * nv[s]) + rq[s] - (e2[s] - d2 * nv[s]);
      ss += v * v;
    }
#pragma unroll
    for (int m = 16; m > 0; m >>= 1) ss += __shfl_xor(ss, m, 32);
    if (lane == 0) {
      float sc = -sqrtf(ss);
      if (i < 10) out[b * 10 + i] = sc;
      else out[2560 + (size_t)b * 512 + (i - 10)] = sc;
    }
  }
}

// ---------------------------------------------------------------------------
// Host orchestration
// ---------------------------------------------------------------------------
extern "C" void kernel_launch(void* const* d_in, const int* in_sizes, int n_in,
                              void* d_out, int out_size, void* d_ws, size_t ws_size,
                              hipStream_t stream) {
  (void)in_sizes; (void)n_in; (void)out_size;
  const float* support = (const float*)d_in[0];
  const float* supneg  = (const float*)d_in[1];
  const float* query   = (const float*)d_in[2];
  const float* negative= (const float*)d_in[3];
  const float* normv   = (const float*)d_in[4];
  const int*   leftc   = (const int*)d_in[5];
  const int*   rightc  = (const int*)d_in[6];
  const float* emb     = (const float*)d_in[7];
  const float* gcn_w   = (const float*)d_in[8];
  const float* gcn_wb  = (const float*)d_in[9];
  const float* gcn_b   = (const float*)d_in[10];
  const float* attn_w  = (const float*)d_in[11];
  /* attn_wb (d_in[12]) is softmax-invariant: skipped */
  const float* gate_w  = (const float*)d_in[13];
  const float* gate_wb = (const float*)d_in[14];
  const float* gate_b  = (const float*)d_in[15];
  const float* Wih0    = (const float*)d_in[16];
  const float* Whh0    = (const float*)d_in[17];
  const float* bih0    = (const float*)d_in[18];
  const float* bhh0    = (const float*)d_in[19];
  const float* Wih1    = (const float*)d_in[20];
  const float* Whh1    = (const float*)d_in[21];
  const float* bih1    = (const float*)d_in[22];
  const float* bhh1    = (const float*)d_in[23];
  const float* out_w   = (const float*)d_in[24];
  const float* out_b   = (const float*)d_in[25];

  uint8_t* base = (uint8_t*)d_ws;
  size_t off = 0;
  auto alloc = [&](size_t bytes) -> void* {
    void* p = base + off;
    off += (bytes + 255) & ~(size_t)255;
    return p;
  };

  // ---- zero-initialized region (bf16 GEMM operands + h/c state) ----
  unsigned short* gcnw  = (unsigned short*)alloc((size_t)112 * 224 * 2);
  unsigned short* x0    = (unsigned short*)alloc((size_t)1280 * 224 * 2);
  unsigned short* wih0p = (unsigned short*)alloc((size_t)2 * 1808 * 224 * 2);
  unsigned short* whh0p = (unsigned short*)alloc((size_t)2 * 1808 * 480 * 2);
  unsigned short* wih1p = (unsigned short*)alloc((size_t)2 * 1808 * 928 * 2);
  unsigned short* whh1p = (unsigned short*)alloc((size_t)2 * 1808 * 480 * 2);
  unsigned short* outwp = (unsigned short*)alloc((size_t)112 * 1824 * 2);
  unsigned short* y0    = (unsigned short*)alloc((size_t)1280 * 928 * 2);
  unsigned short* ctx   = (unsigned short*)alloc((size_t)256 * 1824 * 2);
  unsigned short* hbf   = (unsigned short*)alloc((size_t)2 * 256 * 480 * 2);
  float* cbuf           = (float*)alloc((size_t)2 * 256 * 480 * 4);
  const size_t zero_bytes = off;
  // ---- plain scratch ----
  float* Xpre   = (float*)alloc((size_t)2 * 1280 * 1808 * 4);
  float* G      = (float*)alloc((size_t)2 * 256 * 1808 * 4);
  float* y1     = (float*)alloc((size_t)1280 * 900 * 4);
  float* finals = (float*)alloc((size_t)4 * 256 * 450 * 4);
  float* relb   = (float*)alloc((size_t)256 * 112 * 4);
  if (off > ws_size) return;  // deterministic no-op if workspace too small

  zero_kernel<<<2048, 256, 0, stream>>>((uint4*)base, zero_bytes / 16);

  conv_bf16<<<(100 * 200 + 255) / 256, 256, 0, stream>>>(gcn_w, gcnw, 100, 200, 224);
  conv_bf16<<<(100 * 1800 + 255) / 256, 256, 0, stream>>>(out_w, outwp, 100, 1800, 1824);
  for (int d = 0; d < 2; ++d) {
    conv_bf16<<<(1800 * 200 + 255) / 256, 256, 0, stream>>>(
        Wih0 + (size_t)d * 1800 * 200, wih0p + (size_t)d * 1808 * 224, 1800, 200, 224);
    conv_bf16<<<(1800 * 450 + 255) / 256, 256, 0, stream>>>(
        Whh0 + (size_t)d * 1800 * 450, whh0p + (size_t)d * 1808 * 480, 1800, 450, 480);
    conv_bf16<<<(1800 * 900 + 255) / 256, 256, 0, stream>>>(
        Wih1 + (size_t)d * 1800 * 900, wih1p + (size_t)d * 1808 * 928, 1800, 900, 928);
    conv_bf16<<<(1800 * 450 + 255) / 256, 256, 0, stream>>>(
        Whh1 + (size_t)d * 1800 * 450, whh1p + (size_t)d * 1808 * 480, 1800, 450, 480);
  }

  encoder_kernel<<<2560, 256, 0, stream>>>(leftc, rightc, emb, gcnw, gcn_wb, gcn_b,
                                           attn_w, gate_w, gate_wb, gate_b, x0);

  dim3 blk(256), gX(1280 / 64, 15), gG(256 / 64, 15);

  // ---- layer 0 ----
  for (int d = 0; d < 2; ++d)
    gemm_bf16_wmma<<<gX, blk, 0, stream>>>(
        x0, 224, wih0p + (size_t)d * 1808 * 224, 224,
        Xpre + (size_t)d * 1280 * 1808, 1808, nullptr, 0,
        bih0 + d * 1800, bhh0 + d * 1800, 1808, 1800, 224);
  for (int s = 0; s < 5; ++s) {
    for (int d = 0; d < 2; ++d) {
      int tp = d ? (4 - s) : s;
      gemm_bf16_wmma<<<gG, blk, 0, stream>>>(
          hbf + (size_t)d * 256 * 480, 480, whh0p + (size_t)d * 1808 * 480, 480,
          G + (size_t)d * 256 * 1808, 1808,
          Xpre + (size_t)d * 1280 * 1808 + (size_t)tp * 256 * 1808, 1808,
          nullptr, nullptr, 1808, 1800, 480);
      lstm_cell<<<450, blk, 0, stream>>>(
          G + (size_t)d * 256 * 1808, cbuf + (size_t)d * 256 * 480,
          hbf + (size_t)d * 256 * 480, y0, nullptr,
          (s == 4) ? finals + (size_t)d * 256 * 450 : nullptr, d * 450, tp);
    }
  }
  // reset h (bf16) and c (f32) for layer 1 — contiguous in workspace
  zero_kernel<<<512, 256, 0, stream>>>(
      (uint4*)hbf, ((size_t)2 * 256 * 480 * 2 + (size_t)2 * 256 * 480 * 4) / 16);

  // ---- layer 1 ----
  for (int d = 0; d < 2; ++d)
    gemm_bf16_wmma<<<gX, blk, 0, stream>>>(
        y0, 928, wih1p + (size_t)d * 1808 * 928, 928,
        Xpre + (size_t)d * 1280 * 1808, 1808, nullptr, 0,
        bih1 + d * 1800, bhh1 + d * 1800, 1808, 1800, 928);
  for (int s = 0; s < 5; ++s) {
    for (int d = 0; d < 2; ++d) {
      int tp = d ? (4 - s) : s;
      gemm_bf16_wmma<<<gG, blk, 0, stream>>>(
          hbf + (size_t)d * 256 * 480, 480, whh1p + (size_t)d * 1808 * 480, 480,
          G + (size_t)d * 256 * 1808, 1808,
          Xpre + (size_t)d * 1280 * 1808 + (size_t)tp * 256 * 1808, 1808,
          nullptr, nullptr, 1808, 1800, 480);
      lstm_cell<<<450, blk, 0, stream>>>(
          G + (size_t)d * 256 * 1808, cbuf + (size_t)d * 256 * 480,
          hbf + (size_t)d * 256 * 480, nullptr, y1,
          (s == 4) ? finals + (size_t)(2 + d) * 256 * 450 : nullptr, d * 450, tp);
    }
  }

  attn_kernel<<<256, blk, 0, stream>>>(y1, finals, ctx);
  gemm_bf16_wmma<<<dim3(256 / 64, 1), blk, 0, stream>>>(
      ctx, 1824, outwp, 1824, relb, 112, nullptr, 0, out_b, nullptr, 112, 100, 1824);
  meta_kernel<<<256, blk, 0, stream>>>(support, supneg, query, negative, normv, relb,
                                       (float*)d_out);
}
